// pointer_decoder_31756988187317
// MI455X (gfx1250) — matrix operand order repaired and checked by reading, not compiled, not run on hardware
//
#include <hip/hip_runtime.h>
#include <math.h>

typedef __attribute__((ext_vector_type(16))) __bf16 v16bf;
typedef __attribute__((ext_vector_type(8)))  float  v8f;

#define LL_   2
#define E_    256
#define H_    512
#define V_    50000
#define B_    32
#define S_    1024
#define T_    16
#define OOV_  50
#define EXTV_ (V_ + OOV_)

// ---------------- one-time setup kernels ----------------

__global__ void k_convert_bf16(const float* __restrict__ src, __bf16* __restrict__ dst, int n) {
  int i = blockIdx.x * blockDim.x + threadIdx.x;
  int stride = gridDim.x * blockDim.x;
  for (; i < n; i += stride) dst[i] = (__bf16)src[i];
}

// dst (rows x (Ka+Kb)) = [A | B] converted to bf16, row-major
__global__ void k_concat2_bf16(const float* __restrict__ A, int Ka,
                               const float* __restrict__ Bm, int Kb,
                               __bf16* __restrict__ dst, int rows) {
  int K = Ka + Kb;
  long n = (long)rows * K;
  long i = (long)blockIdx.x * blockDim.x + threadIdx.x;
  long stride = (long)gridDim.x * blockDim.x;
  for (; i < n; i += stride) {
    int r = (int)(i / K);
    int k = (int)(i - (long)r * K);
    float v = (k < Ka) ? A[(size_t)r * Ka + k] : Bm[(size_t)r * Kb + (k - Ka)];
    dst[i] = (__bf16)v;
  }
}

__global__ void k_copy2(const float* __restrict__ a, const float* __restrict__ b,
                        float* __restrict__ da, float* __restrict__ db, int n) {
  int i = blockIdx.x * blockDim.x + threadIdx.x;
  if (i < n) { da[i] = a[i]; db[i] = b[i]; }
}

// ---------------- per-step helper kernels ----------------

__global__ void k_gather_emb(const int* __restrict__ inputs, const float* __restrict__ emb_table,
                             float* __restrict__ emb_t, int t) {
  int i = blockIdx.x * blockDim.x + threadIdx.x;   // b*E + e
  if (i >= B_ * E_) return;
  int b = i >> 8;
  int e = i & (E_ - 1);
  int tok = inputs[b * T_ + t];
  emb_t[i] = emb_table[(size_t)tok * E_ + e];
}

// dst (rows x (Ka+Kb)) = [A | B], fp32
__global__ void k_concat2(const float* __restrict__ A, int Ka,
                          const float* __restrict__ Bm, int Kb,
                          float* __restrict__ dst, int rows) {
  int K = Ka + Kb;
  int n = rows * K;
  int i = blockIdx.x * blockDim.x + threadIdx.x;
  if (i >= n) return;
  int r = i / K;
  int k = i - r * K;
  dst[i] = (k < Ka) ? A[(size_t)r * Ka + k] : Bm[(size_t)r * Kb + (k - Ka)];
}

// -------- generic WMMA GEMM: out(32 x N) = X(32 x K) @ W(N x K)^T + bias --------
// K multiple of 32 (<=1024), N multiple of 16. act: 0 = none, 1 = tanh.
__global__ __launch_bounds__(256) void k_gemm_wmma(
    const float* __restrict__ X, int K,
    const __bf16* __restrict__ W, const float* __restrict__ bias,
    int N, float* __restrict__ out, int act) {
  // A fragments in contiguous k order (k = 16*half + i): [mtile][kchunk][lane] -> 16 bf16 each
  __shared__ v16bf ldsA[2 * 32 * 32];                // up to KC=32 (K=1024): 64KB
  int tid = threadIdx.x;
  int KC = K >> 5;
  __bf16* ldsAh = (__bf16*)ldsA;
  int totalHalves = 2 * KC * 512;                    // always a multiple of 256
  for (int idx = tid; idx < totalHalves; idx += 256) {
    int i    = idx & 15;                             // half index within fragment
    int lane = (idx >> 4) & 31;
    int ch   = idx >> 9;                             // mt*KC + kc
    int kc   = ch % KC;
    int mt   = ch / KC;
    int m = mt * 16 + (lane & 15);
    int Kk = (lane < 16) ? i
                         : i + 16;
    int k = kc * 32 + Kk;
    ldsAh[idx] = (__bf16)X[(size_t)m * K + k];
  }
  __syncthreads();

  int lane = tid & 31;
  int wid  = tid >> 5;
  int nt = blockIdx.x * 8 + wid;                     // one 16-wide N tile per wave
  int ntiles = N >> 4;
  if (nt >= ntiles) return;                          // wave-uniform: EXEC stays all-1s

  int n = nt * 16 + (lane & 15);                     // B layout: lanes 0-15 K 0..15, lanes 16-31 K 16..31
  const __bf16* bp = W + (size_t)n * K + ((lane >> 4) ? 16 : 0);

  v8f acc0 = {};
  v8f acc1 = {};
#pragma unroll 2
  for (int kc = 0; kc < KC; ++kc) {
    v16bf bfrag = *(const v16bf*)(bp + kc * 32);     // 16 contiguous K values
    v16bf a0 = ldsA[kc * 32 + lane];
    v16bf a1 = ldsA[(KC + kc) * 32 + lane];
    acc0 = __builtin_amdgcn_wmma_f32_16x16x32_bf16(false, a0, false, bfrag,
                                                   (short)0, acc0, false, false);
    acc1 = __builtin_amdgcn_wmma_f32_16x16x32_bf16(false, a1, false, bfrag,
                                                   (short)0, acc1, false, false);
  }
  float bv = bias[n];
  int mbase = (lane >> 4) ? 8 : 0;                   // C/D layout: VGPR r -> row r (+8 upper lanes)
#pragma unroll
  for (int r = 0; r < 8; ++r) {
    int m = mbase + r;
    float v0 = acc0[r] + bv;
    float v1 = acc1[r] + bv;
    if (act == 1) { v0 = tanhf(v0); v1 = tanhf(v1); }
    out[(size_t)m * N + n]        = v0;
    out[(size_t)(m + 16) * N + n] = v1;
  }
}

// LSTM elementwise: gates (32 x 4H, order i,f,g,o) -> h', c'
__global__ void k_lstm_act(const float* __restrict__ gates, const float* __restrict__ cprev,
                           float* __restrict__ hnew, float* __restrict__ cnew) {
  int i = blockIdx.x * blockDim.x + threadIdx.x;     // b*H + u
  if (i >= B_ * H_) return;
  int b = i >> 9;
  int u = i & (H_ - 1);
  const float* g = gates + (size_t)b * (4 * H_);
  float ig = 1.f / (1.f + __expf(-g[u]));
  float fg = 1.f / (1.f + __expf(-g[H_ + u]));
  float gg = tanhf(g[2 * H_ + u]);
  float og = 1.f / (1.f + __expf(-g[3 * H_ + u]));
  float c2 = fg * cprev[i] + ig * gg;
  cnew[i] = c2;
  hnew[i] = og * tanhf(c2);
}

// one wave per (b, s) row; bf16 contexts stream, fp32 accumulate, shuffle reduce
__global__ void k_scores(const __bf16* __restrict__ ctx_bf, const float* __restrict__ gamma,
                         const int* __restrict__ enc_mask, float* __restrict__ scores) {
  int row  = blockIdx.x * 8 + (threadIdx.x >> 5);    // over B_*S_ rows
  int lane = threadIdx.x & 31;
  if (row >= B_ * S_) return;
  int b = row >> 10;
  const __bf16* cx = ctx_bf + (size_t)row * H_;
  const float* gb = gamma + (size_t)b * H_;
  float acc = 0.f;
  for (int k = lane; k < H_; k += 32) acc += (float)cx[k] * gb[k];
#pragma unroll
  for (int m = 16; m; m >>= 1) acc += __shfl_xor(acc, m, 32);
  if (lane == 0) scores[row] = (enc_mask[row] > 0) ? acc : -1e9f;
}

__global__ void k_attn_softmax(const float* __restrict__ scores, float* __restrict__ attn) {
  __shared__ float red[256];
  int b = blockIdx.x, tid = threadIdx.x;
  const float* sr = scores + (size_t)b * S_;
  float m = -1e30f;
  for (int s = tid; s < S_; s += 256) m = fmaxf(m, sr[s]);
  red[tid] = m; __syncthreads();
  for (int w = 128; w; w >>= 1) { if (tid < w) red[tid] = fmaxf(red[tid], red[tid + w]); __syncthreads(); }
  m = red[0]; __syncthreads();
  float sum = 0.f;
  for (int s = tid; s < S_; s += 256) sum += __expf(sr[s] - m);
  red[tid] = sum; __syncthreads();
  for (int w = 128; w; w >>= 1) { if (tid < w) red[tid] += red[tid + w]; __syncthreads(); }
  float inv = 1.f / red[0];
  for (int s = tid; s < S_; s += 256) attn[(size_t)b * S_ + s] = __expf(sr[s] - m) * inv;
}

__global__ void k_ct(const float* __restrict__ attn, const __bf16* __restrict__ ctx_bf,
                     float* __restrict__ ct) {
  int i = blockIdx.x * blockDim.x + threadIdx.x;     // b*H + u
  if (i >= B_ * H_) return;
  int b = i >> 9;
  int u = i & (H_ - 1);
  const float* ab = attn + (size_t)b * S_;
  const __bf16* cb = ctx_bf + (size_t)b * S_ * H_ + u;
  float acc = 0.f;
  for (int s = 0; s < S_; ++s) acc += ab[s] * (float)cb[(size_t)s * H_];
  ct[i] = acc;
}

// one wave per batch row: pg_in = [h1, c1, text_out, emb]
__global__ void k_pgen(const float* __restrict__ h1, const float* __restrict__ c1,
                       const float* __restrict__ to, const float* __restrict__ emb,
                       const float* __restrict__ pgW, const float* __restrict__ pgb,
                       float* __restrict__ pgen) {
  int b = blockIdx.x, lane = threadIdx.x;
  float acc = 0.f;
  for (int j = lane; j < E_ + 3 * H_; j += 32) {
    float v;
    if (j < H_)          v = h1[(size_t)b * H_ + j];
    else if (j < 2 * H_) v = c1[(size_t)b * H_ + j - H_];
    else if (j < 3 * H_) v = to[(size_t)b * H_ + j - 2 * H_];
    else                 v = emb[(size_t)b * E_ + j - 3 * H_];
    acc += v * pgW[j];
  }
#pragma unroll
  for (int m = 16; m; m >>= 1) acc += __shfl_xor(acc, m, 32);
  if (lane == 0) pgen[b] = 1.f / (1.f + __expf(-(acc + pgb[0])));
}

__global__ void k_vocab_reduce(const float* __restrict__ logits,
                               float* __restrict__ rmax, float* __restrict__ rsum) {
  __shared__ float red[256];
  int b = blockIdx.x, tid = threadIdx.x;
  const float* lr = logits + (size_t)b * V_;
  float m = -1e30f;
  for (int v = tid; v < V_; v += 256) m = fmaxf(m, lr[v]);
  red[tid] = m; __syncthreads();
  for (int w = 128; w; w >>= 1) { if (tid < w) red[tid] = fmaxf(red[tid], red[tid + w]); __syncthreads(); }
  m = red[0]; __syncthreads();
  float s = 0.f;
  for (int v = tid; v < V_; v += 256) s += __expf(lr[v] - m);
  red[tid] = s; __syncthreads();
  for (int w = 128; w; w >>= 1) { if (tid < w) red[tid] += red[tid + w]; __syncthreads(); }
  if (tid == 0) { rmax[b] = m; rsum[b] = red[0]; }
}

__global__ void k_write_out(const float* __restrict__ logits, const float* __restrict__ rmax,
                            const float* __restrict__ rsum, const float* __restrict__ pgen,
                            float* __restrict__ out, int t) {
  int i = blockIdx.x * blockDim.x + threadIdx.x;
  if (i >= B_ * EXTV_) return;
  int b = i / EXTV_;
  int v = i - b * EXTV_;
  float val = 0.f;
  if (v < V_)
    val = __expf(logits[(size_t)b * V_ + v] - rmax[b]) * (pgen[b] / rsum[b]);
  out[((size_t)b * T_ + t) * EXTV_ + v] = val;
}

__global__ void k_scatter(const float* __restrict__ attn, const float* __restrict__ pgen,
                          const int* __restrict__ title_index, const int* __restrict__ extend_vocab,
                          float* __restrict__ out, int t) {
  int i = blockIdx.x * blockDim.x + threadIdx.x;
  if (i >= B_ * S_) return;
  int b = i >> 10;
  int s = i & (S_ - 1);
  float tm = (s == title_index[b]) ? 0.f : 1.f;
  float w = (1.f - pgen[b]) * attn[i] * tm + 1e-12f;
  atomicAdd(&out[((size_t)b * T_ + t) * EXTV_ + extend_vocab[i]], w);
}

// ---------------- host orchestration ----------------

extern "C" void kernel_launch(void* const* d_in, const int* in_sizes, int n_in,
                              void* d_out, int out_size, void* d_ws, size_t ws_size,
                              hipStream_t stream) {
  (void)in_sizes; (void)n_in; (void)out_size; (void)ws_size;
  const int*   inputs    = (const int*)d_in[0];
  const float* h0        = (const float*)d_in[1];
  const float* c0        = (const float*)d_in[2];
  const float* contexts  = (const float*)d_in[3];
  const int*   enc_mask  = (const int*)d_in[4];
  const int*   title_ix  = (const int*)d_in[5];
  const int*   ext_vocab = (const int*)d_in[7];
  const float* emb_table = (const float*)d_in[8];
  const float* W_ih0     = (const float*)d_in[9];
  const float* W_hh0     = (const float*)d_in[10];
  const float* b0        = (const float*)d_in[11];
  const float* W_ih1     = (const float*)d_in[12];
  const float* W_hh1     = (const float*)d_in[13];
  const float* b1        = (const float*)d_in[14];
  const float* att_Win   = (const float*)d_in[15];
  const float* att_bin   = (const float*)d_in[16];
  const float* att_Wout  = (const float*)d_in[17];
  const float* att_bout  = (const float*)d_in[18];
  const float* lin_W     = (const float*)d_in[19];
  const float* lin_b     = (const float*)d_in[20];
  const float* pg_W      = (const float*)d_in[21];
  const float* pg_b      = (const float*)d_in[22];
  float* out = (float*)d_out;

  char* ws = (char*)d_ws;
  size_t off = 0;
  auto walloc = [&](size_t bytes) -> void* {
    void* p = ws + off;
    off = (off + bytes + 255) & ~(size_t)255;
    return p;
  };
  __bf16* Wlinbf  = (__bf16*)walloc((size_t)V_ * H_ * sizeof(__bf16));              // 50000 x 512
  __bf16* ctx_bf  = (__bf16*)walloc((size_t)B_ * S_ * H_ * sizeof(__bf16));         // 32 x 1024 x 512
  __bf16* W0bf    = (__bf16*)walloc((size_t)(4 * H_) * (E_ + H_) * sizeof(__bf16)); // 2048 x 768
  __bf16* W1bf    = (__bf16*)walloc((size_t)(4 * H_) * (2 * H_) * sizeof(__bf16));  // 2048 x 1024
  __bf16* Winbf   = (__bf16*)walloc((size_t)H_ * H_ * sizeof(__bf16));              // 512 x 512
  __bf16* Woutbf  = (__bf16*)walloc((size_t)H_ * (2 * H_) * sizeof(__bf16));        // 512 x 1024
  float* logits   = (float*)walloc((size_t)B_ * V_ * sizeof(float));
  float* gates    = (float*)walloc((size_t)B_ * 4 * H_ * sizeof(float));
  float* xcat0    = (float*)walloc((size_t)B_ * (E_ + H_) * sizeof(float));         // [emb | h0prev]
  float* xcat1    = (float*)walloc((size_t)B_ * (2 * H_) * sizeof(float));          // [h0new | h1prev]
  float* xcat2    = (float*)walloc((size_t)B_ * (2 * H_) * sizeof(float));          // [ct | h1new]
  float* hbuf0    = (float*)walloc((size_t)LL_ * B_ * H_ * sizeof(float));
  float* cbuf0    = (float*)walloc((size_t)LL_ * B_ * H_ * sizeof(float));
  float* hbuf1    = (float*)walloc((size_t)LL_ * B_ * H_ * sizeof(float));
  float* cbuf1    = (float*)walloc((size_t)LL_ * B_ * H_ * sizeof(float));
  float* emb_t    = (float*)walloc((size_t)B_ * E_ * sizeof(float));
  float* gamma    = (float*)walloc((size_t)B_ * H_ * sizeof(float));
  float* scores   = (float*)walloc((size_t)B_ * S_ * sizeof(float));
  float* attn     = (float*)walloc((size_t)B_ * S_ * sizeof(float));
  float* ctb      = (float*)walloc((size_t)B_ * H_ * sizeof(float));
  float* text_out = (float*)walloc((size_t)B_ * H_ * sizeof(float));
  float* pgen     = (float*)walloc((size_t)B_ * sizeof(float));
  float* rmax     = (float*)walloc((size_t)B_ * sizeof(float));
  float* rsum     = (float*)walloc((size_t)B_ * sizeof(float));

  // hoisted out of the time loop: bf16 conversions (weights + contexts), state init
  k_convert_bf16<<<2048, 256, 0, stream>>>(lin_W, Wlinbf, V_ * H_);
  k_convert_bf16<<<2048, 256, 0, stream>>>(contexts, ctx_bf, B_ * S_ * H_);
  k_concat2_bf16<<<2048, 256, 0, stream>>>(W_ih0, E_, W_hh0, H_, W0bf, 4 * H_);
  k_concat2_bf16<<<2048, 256, 0, stream>>>(W_ih1, H_, W_hh1, H_, W1bf, 4 * H_);
  k_convert_bf16<<<512, 256, 0, stream>>>(att_Win, Winbf, H_ * H_);
  k_convert_bf16<<<1024, 256, 0, stream>>>(att_Wout, Woutbf, H_ * 2 * H_);
  k_copy2<<<(LL_ * B_ * H_ + 255) / 256, 256, 0, stream>>>(h0, c0, hbuf0, cbuf0, LL_ * B_ * H_);

  float* hP[2] = {hbuf0, hbuf1};
  float* cP[2] = {cbuf0, cbuf1};
  for (int t = 0; t < T_; ++t) {
    float* hprev = hP[t & 1];
    float* cprev = cP[t & 1];
    float* hnew  = hP[(t & 1) ^ 1];
    float* cnew  = cP[(t & 1) ^ 1];
    float* h1new = hnew + B_ * H_;
    float* c1new = cnew + B_ * H_;

    // LSTM layer 0: gates = [emb | h0prev] @ [Wih0|Whh0]^T + b0
    k_gather_emb<<<(B_ * E_) / 256, 256, 0, stream>>>(inputs, emb_table, emb_t, t);
    k_concat2<<<(B_ * (E_ + H_) + 255) / 256, 256, 0, stream>>>(emb_t, E_, hprev, H_, xcat0, B_);
    k_gemm_wmma<<<(4 * H_ / 16 + 7) / 8, 256, 0, stream>>>(xcat0, E_ + H_, W0bf, b0, 4 * H_, gates, 0);
    k_lstm_act<<<(B_ * H_) / 256, 256, 0, stream>>>(gates, cprev, hnew, cnew);

    // LSTM layer 1: gates = [h0new | h1prev] @ [Wih1|Whh1]^T + b1
    k_concat2<<<(B_ * 2 * H_ + 255) / 256, 256, 0, stream>>>(hnew, H_, hprev + B_ * H_, H_, xcat1, B_);
    k_gemm_wmma<<<(4 * H_ / 16 + 7) / 8, 256, 0, stream>>>(xcat1, 2 * H_, W1bf, b1, 4 * H_, gates, 0);
    k_lstm_act<<<(B_ * H_) / 256, 256, 0, stream>>>(gates, cprev + B_ * H_, h1new, c1new);

    // attention
    k_gemm_wmma<<<(H_ / 16 + 7) / 8, 256, 0, stream>>>(h1new, H_, Winbf, att_bin, H_, gamma, 0);
    k_scores<<<(B_ * S_) / 8, 256, 0, stream>>>(ctx_bf, gamma, enc_mask, scores);
    k_attn_softmax<<<B_, 256, 0, stream>>>(scores, attn);
    k_ct<<<(B_ * H_) / 256, 256, 0, stream>>>(attn, ctx_bf, ctb);
    k_concat2<<<(B_ * 2 * H_ + 255) / 256, 256, 0, stream>>>(ctb, H_, h1new, H_, xcat2, B_);
    k_gemm_wmma<<<(H_ / 16 + 7) / 8, 256, 0, stream>>>(xcat2, 2 * H_, Woutbf, att_bout, H_, text_out, 1);

    // pointer-generator gate + vocab distribution
    k_pgen<<<B_, 32, 0, stream>>>(h1new, c1new, text_out, emb_t, pg_W, pg_b, pgen);
    k_gemm_wmma<<<(V_ / 16 + 7) / 8, 256, 0, stream>>>(text_out, H_, Wlinbf, lin_b, V_, logits, 0);
    k_vocab_reduce<<<B_, 256, 0, stream>>>(logits, rmax, rsum);
    k_write_out<<<(B_ * EXTV_ + 255) / 256, 256, 0, stream>>>(logits, rmax, rsum, pgen, out, t);
    k_scatter<<<(B_ * S_) / 256, 256, 0, stream>>>(attn, pgen, title_ix, ext_vocab, out, t);
  }
}
